// GNNReRanker_27419071217926
// MI455X (gfx1250) — matrix operand order repaired and checked by reading, not compiled, not run on hardware
//
#include <hip/hip_runtime.h>
#include <hip/hip_bf16.h>

#define NN 50000
#define DIN 512
#define HID 128
#define LDSS 40   // padded LDS row stride (ushorts) to avoid bank conflicts

typedef __attribute__((ext_vector_type(16))) __bf16 v16bf;
typedef __attribute__((ext_vector_type(8)))  float  v8f;

union FragB { v16bf v; uint4 q[2]; };

__device__ __forceinline__ unsigned short f32_to_bf16_rne(float f) {
  unsigned int u = __float_as_uint(f);
  unsigned int r = u + 0x7FFFu + ((u >> 16) & 1u);
  return (unsigned short)(r >> 16);
}
__device__ __forceinline__ float bfus_to_f32(unsigned short h) {
  return __uint_as_float(((unsigned int)h) << 16);
}

// ---------------------------------------------------------------------------
// GEMM: H[50000,128] = X[50000,512] @ W1[512,128], fp32 in/out,
// bf16 WMMA with hi/lo error compensation (fp32-class accuracy).
// Software-pipelined: next K-chunk's global loads overlap current WMMAs.
// ---------------------------------------------------------------------------
__global__ __launch_bounds__(256)
void gemm_xw1(const float* __restrict__ X, const float* __restrict__ W1,
              float* __restrict__ H) {
  __shared__ unsigned short AsH[128 * LDSS];
  __shared__ unsigned short AsL[128 * LDSS];
  __shared__ unsigned short BsH[128 * LDSS];
  __shared__ unsigned short BsL[128 * LDSS];

  const int tid  = threadIdx.x;
  const int wave = tid >> 5;
  const int lane = tid & 31;
  const int half = lane >> 4;   // lanes 16-31 hold the other K/M half
  const int r16  = lane & 15;
  const int row0 = blockIdx.x * 128;

  v8f acc[8];
  const v8f zero = {0.f, 0.f, 0.f, 0.f, 0.f, 0.f, 0.f, 0.f};
#pragma unroll
  for (int n = 0; n < 8; ++n) acc[n] = zero;

  float4 ax[4];   // staged A chunk (this thread's share)
  float  wv[16];  // staged B chunk (this thread's share)

  auto load_global = [&](int k0) {
#pragma unroll
    for (int j = 0; j < 4; ++j) {
      int linear = tid + j * 256;          // 1024 float4s = 128 rows x 32 k
      int r  = linear >> 3;
      int c4 = linear & 7;
      int grow = row0 + r;
      int gr = grow < NN ? grow : (NN - 1);   // clamp; store is guarded later
      ax[j] = *reinterpret_cast<const float4*>(&X[(size_t)gr * DIN + k0 + c4 * 4]);
    }
#pragma unroll
    for (int j = 0; j < 16; ++j) {
      int linear = tid + j * 256;          // 4096 elements = 32 k x 128 cols
      int k   = linear >> 7;
      int col = linear & 127;
      wv[j] = W1[(size_t)(k0 + k) * HID + col];
    }
  };

  auto stage_lds = [&]() {
#pragma unroll
    for (int j = 0; j < 4; ++j) {
      int linear = tid + j * 256;
      int r  = linear >> 3;
      int c4 = linear & 7;
      float vx[4] = {ax[j].x, ax[j].y, ax[j].z, ax[j].w};
#pragma unroll
      for (int i = 0; i < 4; ++i) {
        unsigned short hi = f32_to_bf16_rne(vx[i]);
        float lo = vx[i] - bfus_to_f32(hi);
        AsH[r * LDSS + c4 * 4 + i] = hi;
        AsL[r * LDSS + c4 * 4 + i] = f32_to_bf16_rne(lo);
      }
    }
#pragma unroll
    for (int j = 0; j < 16; ++j) {
      int linear = tid + j * 256;
      int k   = linear >> 7;
      int col = linear & 127;
      unsigned short hi = f32_to_bf16_rne(wv[j]);
      float lo = wv[j] - bfus_to_f32(hi);
      BsH[col * LDSS + k] = hi;
      BsL[col * LDSS + k] = f32_to_bf16_rne(lo);
    }
  };

  load_global(0);

  for (int k0 = 0; k0 < DIN; k0 += 32) {
    stage_lds();
    __syncthreads();

    // Kick off next chunk's global loads; they retire during the WMMA block.
    if (k0 + 32 < DIN) {
      // warm L2/WGP$ two chunks ahead for the X stream (global_prefetch_b8)
      if (k0 + 64 < DIN) {
        int linear = tid;
        int r  = linear >> 3;
        int c4 = linear & 7;
        int grow = row0 + r;
        int gr = grow < NN ? grow : (NN - 1);
        __builtin_prefetch(&X[(size_t)gr * DIN + k0 + 64 + c4 * 4], 0, 3);
      }
      load_global(k0 + 32);
    }

    // A fragment: 16 contiguous k at [arow][half*16] (matches B)
    const int arow = wave * 16 + r16;
    FragB ah, al;
    ah.q[0] = *reinterpret_cast<const uint4*>(&AsH[arow * LDSS + half * 16]);
    ah.q[1] = *reinterpret_cast<const uint4*>(&AsH[arow * LDSS + half * 16 + 8]);
    al.q[0] = *reinterpret_cast<const uint4*>(&AsL[arow * LDSS + half * 16]);
    al.q[1] = *reinterpret_cast<const uint4*>(&AsL[arow * LDSS + half * 16 + 8]);

#pragma unroll
    for (int n = 0; n < 8; ++n) {
      int col = n * 16 + r16;
      FragB bh, bl;   // B fragment: 16 contiguous k at [col][half*16]
      bh.q[0] = *reinterpret_cast<const uint4*>(&BsH[col * LDSS + half * 16]);
      bh.q[1] = *reinterpret_cast<const uint4*>(&BsH[col * LDSS + half * 16 + 8]);
      bl.q[0] = *reinterpret_cast<const uint4*>(&BsL[col * LDSS + half * 16]);
      bl.q[1] = *reinterpret_cast<const uint4*>(&BsL[col * LDSS + half * 16 + 8]);
      acc[n] = __builtin_amdgcn_wmma_f32_16x16x32_bf16(
          false, ah.v, false, bh.v, (short)0, acc[n], false, false);
      acc[n] = __builtin_amdgcn_wmma_f32_16x16x32_bf16(
          false, ah.v, false, bl.v, (short)0, acc[n], false, false);
      acc[n] = __builtin_amdgcn_wmma_f32_16x16x32_bf16(
          false, al.v, false, bh.v, (short)0, acc[n], false, false);
    }
    __syncthreads();
  }

  // C layout: VGPR g -> M = half*8 + g, N = r16
#pragma unroll
  for (int n = 0; n < 8; ++n) {
#pragma unroll
    for (int g = 0; g < 8; ++g) {
      int row = row0 + wave * 16 + half * 8 + g;
      int col = n * 16 + r16;
      if (row < NN) H[(size_t)row * HID + col] = acc[n][g];
    }
  }
}

// ---------------------------------------------------------------------------
// Graph passes
// ---------------------------------------------------------------------------
__global__ void init_cnt(unsigned int* cnt) {
  int i = blockIdx.x * blockDim.x + threadIdx.x;
  if (i < NN) cnt[i] = 0u;
}

__global__ void count_deg(const int* __restrict__ dst, unsigned int* cnt, int E) {
  int i = blockIdx.x * blockDim.x + threadIdx.x;
  if (i < E) atomicAdd(&cnt[dst[i]], 1u);
}

__global__ void make_dinv(const unsigned int* __restrict__ cnt, float* dinv) {
  int i = blockIdx.x * blockDim.x + threadIdx.x;
  if (i < NN) dinv[i] = rsqrtf((float)(cnt[i] + 1u));   // +1 self-loop
}

// agg[i][:] = dinv[i]^2 * H[i][:]   (self-loop term; also zero-initializes agg)
__global__ void agg_init(const float* __restrict__ H, const float* __restrict__ dinv,
                         float* __restrict__ agg) {
  int t = blockIdx.x * blockDim.x + threadIdx.x;
  if (t < NN * HID) {
    int i = t >> 7;
    float d = dinv[i];
    agg[t] = d * d * H[t];
  }
}

// one wave per edge: agg[dst] += dinv[s]*dinv[d] * H[src]  (128 f32, fp atomics)
__global__ __launch_bounds__(256)
void edge_feat(const int* __restrict__ src, const int* __restrict__ dst,
               const float* __restrict__ dinv, const float* __restrict__ H,
               float* __restrict__ agg, int E) {
  int gw = (blockIdx.x * 256 + threadIdx.x) >> 5;
  int lane = threadIdx.x & 31;
  if (gw >= E) return;

  // Warm caches for a future edge's gathered row (global_prefetch_b8);
  // each lane covers 16B of the 512B row.
  int gw2 = gw + 128;
  if (gw2 < E) {
    int s2 = src[gw2];
    __builtin_prefetch(&H[(size_t)s2 * HID + lane * 4], 0, 3);
  }

  int s = src[gw], d = dst[gw];
  float nrm = dinv[s] * dinv[d];
  const float4 v =
      *reinterpret_cast<const float4*>(&H[(size_t)s * HID + lane * 4]);
  float* base = &agg[(size_t)d * HID + lane * 4];
  __hip_atomic_fetch_add(&base[0], nrm * v.x, __ATOMIC_RELAXED, __HIP_MEMORY_SCOPE_AGENT);
  __hip_atomic_fetch_add(&base[1], nrm * v.y, __ATOMIC_RELAXED, __HIP_MEMORY_SCOPE_AGENT);
  __hip_atomic_fetch_add(&base[2], nrm * v.z, __ATOMIC_RELAXED, __HIP_MEMORY_SCOPE_AGENT);
  __hip_atomic_fetch_add(&base[3], nrm * v.w, __ATOMIC_RELAXED, __HIP_MEMORY_SCOPE_AGENT);
}

// t[i] = sum_f relu(agg[i][f] + b1[f]) * W2[f]   (one wave per node)
__global__ __launch_bounds__(256)
void relu_dot(const float* __restrict__ agg, const float* __restrict__ b1,
              const float* __restrict__ W2, float* __restrict__ t) {
  int wave = threadIdx.x >> 5, lane = threadIdx.x & 31;
  int node = blockIdx.x * 8 + wave;
  if (node >= NN) return;
  float acc = 0.f;
#pragma unroll
  for (int j = 0; j < 4; ++j) {
    int f = lane + j * 32;
    float h = agg[(size_t)node * HID + f] + b1[f];
    h = h > 0.f ? h : 0.f;
    acc += h * W2[f];
  }
  for (int off = 16; off > 0; off >>= 1) acc += __shfl_xor(acc, off, 32);
  if (lane == 0) t[node] = acc;
}

// out[i] = b2 + dinv[i]^2 * t[i]   (also overwrites poisoned d_out)
__global__ void out_init(const float* __restrict__ t, const float* __restrict__ dinv,
                         const float* __restrict__ b2, float* __restrict__ out) {
  int i = blockIdx.x * blockDim.x + threadIdx.x;
  if (i < NN) {
    float d = dinv[i];
    out[i] = b2[0] + d * d * t[i];
  }
}

__global__ void edge_scalar(const int* __restrict__ src, const int* __restrict__ dst,
                            const float* __restrict__ dinv, const float* __restrict__ t,
                            float* __restrict__ out, int E) {
  int i = blockIdx.x * blockDim.x + threadIdx.x;
  if (i < E) {
    int s = src[i], d = dst[i];
    __hip_atomic_fetch_add(&out[d], dinv[s] * dinv[d] * t[s],
                           __ATOMIC_RELAXED, __HIP_MEMORY_SCOPE_AGENT);
  }
}

// ---------------------------------------------------------------------------
extern "C" void kernel_launch(void* const* d_in, const int* in_sizes, int n_in,
                              void* d_out, int out_size, void* d_ws, size_t ws_size,
                              hipStream_t stream) {
  const float* X  = (const float*)d_in[0];
  const int*   EI = (const int*)d_in[1];
  const float* W1 = (const float*)d_in[2];
  const float* b1 = (const float*)d_in[3];
  const float* W2 = (const float*)d_in[4];
  const float* b2 = (const float*)d_in[5];
  float* out = (float*)d_out;

  const int E = in_sizes[1] / 2;
  const int* src = EI;
  const int* dst = EI + E;

  float* ws = (float*)d_ws;
  unsigned int* cnt = (unsigned int*)ws;          // 50000 u32
  float* dinv = ws + 65536;                       // 50000 f32
  float* tbuf = ws + 131072;                      // 50000 f32
  float* H    = ws + 196608;                      // 50000*128 f32
  float* agg  = H + (size_t)NN * HID;             // 50000*128 f32  (~52 MB total)

  init_cnt   <<<(NN + 255) / 256, 256, 0, stream>>>(cnt);
  count_deg  <<<(E + 255) / 256, 256, 0, stream>>>(dst, cnt, E);
  make_dinv  <<<(NN + 255) / 256, 256, 0, stream>>>(cnt, dinv);
  gemm_xw1   <<<(NN + 127) / 128, 256, 0, stream>>>(X, W1, H);
  agg_init   <<<(NN * HID + 255) / 256, 256, 0, stream>>>(H, dinv, agg);
  edge_feat  <<<(E + 7) / 8, 256, 0, stream>>>(src, dst, dinv, H, agg, E);
  relu_dot   <<<(NN + 7) / 8, 256, 0, stream>>>(agg, b1, W2, tbuf);
  out_init   <<<(NN + 255) / 256, 256, 0, stream>>>(tbuf, dinv, b2, out);
  edge_scalar<<<(E + 255) / 256, 256, 0, stream>>>(src, dst, dinv, tbuf, out, E);
}
